// EFEPlanner_16638703305229
// MI455X (gfx1250) — matrix operand order repaired and checked by reading, not compiled, not run on hardware
//
#include <hip/hip_runtime.h>

// ---------------------------------------------------------------------------
// EFE planner rollout for MI455X (gfx1250), bf16 WMMA path.
//
// Roofline: ~423 GFLOP total, weights ~10MB bf16 -> resident in 192MB L2.
// Compute-bound => v_wmma_f32_16x16x32_bf16 everywhere. The one-hot action
// concat is algebraically a row-lookup of Wx (rows S..S+A-1) => folded into a
// per-row bias, making every GEMM K a multiple of 32.
//
// Wave work assignment: each wave owns a unique column stripe and BOTH 16-row
// tiles, so each B-fragment (1KB) feeds two WMMAs and every weight tile is
// loaded exactly once per WG per step (A-fragments stream from LDS).
// Noise for step t is staged global->LDS with async b128 copies overlapping
// phase A (ASYNCcnt), instead of scattered scalar loads in the epilogue.
// ---------------------------------------------------------------------------

#define A_DIM   16
#define NSAMP   256
#define BATCH   4096
#define T_HOR   10
#define HS      1024
#define SS      256
#define FEATN   1280
#define RHN     512

typedef __bf16 bf16_t;
typedef __attribute__((ext_vector_type(16))) __bf16 v16bf;
typedef __attribute__((ext_vector_type(8)))  float  v8f;

__device__ __forceinline__ v8f wmma_bf16(v16bf a, v16bf b, v8f c) {
  // D = A(16x32) * B(32x16) + C(16x16 f32)
  return __builtin_amdgcn_wmma_f32_16x16x32_bf16(
      /*neg_a=*/false, a, /*neg_b=*/false, b,
      /*c_mod=*/(short)0, c, /*reuse_a=*/false, /*reuse_b=*/false);
}

// A-fragment from row-major bf16 LDS matrix [rows][ld].
// ISA 7.12.2 (16-bit A 16x32): lane<16 holds M=lane, K = {k0..k0+7, k0+16..k0+23};
// lane>=16 holds M=lane-16, K = {k0+8..k0+15, k0+24..k0+31}.
__device__ __forceinline__ v16bf load_A_lds(const bf16_t* base, int ld,
                                            int m0, int k0, int lane) {
  const int m  = m0 + (lane & 15);
  const int kh = (lane >> 4) << 3;            // 0 or 8
  const bf16_t* p = base + m * ld + k0 + kh;
  v16bf a;
  reinterpret_cast<float4*>(&a)[0] = *reinterpret_cast<const float4*>(p);      // K+0..7
  reinterpret_cast<float4*>(&a)[1] = *reinterpret_cast<const float4*>(p + 16); // K+16..23
  return a;
}

// B-fragment from pre-packed global weights. Packed layout: tile(kt,nt) of
// 32x16, block index nt*KT+kt (K-contiguous stream), 32 bytes per lane.
// Assumed B layout: lane n = lane&15, K = (lane>>4)*8 + (j&7) + (j>>3)*16, j=0..15.
__device__ __forceinline__ v16bf load_B_g(const bf16_t* packed, int kt, int nt,
                                          int KT, int lane) {
  const bf16_t* p = packed + ((((size_t)nt * KT + kt) << 5) + lane) * 16;
  v16bf b;
  reinterpret_cast<float4*>(&b)[0] = reinterpret_cast<const float4*>(p)[0];
  reinterpret_cast<float4*>(&b)[1] = reinterpret_cast<const float4*>(p)[1];
  return b;
}

// Async global -> LDS copy of 16 bytes per lane (tracked by ASYNCcnt).
__device__ __forceinline__ void async_g2l_b128(uint32_t lds_off, uint64_t gaddr) {
  asm volatile("global_load_async_to_lds_b128 %0, %1, off"
               :: "v"(lds_off), "v"(gaddr) : "memory");
}
__device__ __forceinline__ void wait_asynccnt0() {
  asm volatile("s_wait_asynccnt 0x0" ::: "memory");
}

// ---------------------------------------------------------------------------
// Prologue: pack fp32 row-major W[K][N] -> bf16 WMMA-B tile order.
// ---------------------------------------------------------------------------
__global__ void pack_b_kernel(const float* __restrict__ W, int K, int N,
                              bf16_t* __restrict__ out) {
  const int KT = K >> 5, NT = N >> 4;
  const int id = blockIdx.x * blockDim.x + threadIdx.x;
  const int total = KT * NT * 32;
  if (id >= total) return;
  const int lane = id & 31, block = id >> 5;
  const int kt = block % KT, nt = block / KT;
  const int n = nt * 16 + (lane & 15);
  const int kbase = kt * 32 + ((lane >> 4) << 3);
  bf16_t* dst = out + (size_t)block * 512 + lane * 16;
#pragma unroll
  for (int j = 0; j < 16; ++j)
    dst[j] = (bf16_t)W[(size_t)(kbase + (j & 7) + ((j >> 3) << 4)) * N + n];
}

// Copy Wx rows [S, S+A) as fp32 action-bias table (A_DIM x 3H).
__global__ void copy_wxa_kernel(const float* __restrict__ Wx,
                                float* __restrict__ out) {
  const int id = blockIdx.x * blockDim.x + threadIdx.x;
  if (id < A_DIM * 3 * HS) out[id] = Wx[(size_t)SS * 3 * HS + id];
}

// ---------------------------------------------------------------------------
// Main rollout: 128 WGs x 256 thr (8 wave32). 32 rows/WG held across T steps.
// ---------------------------------------------------------------------------
__global__ __launch_bounds__(256, 1) void efe_rollout(
    const float* __restrict__ h0, const float* __restrict__ s0,
    const float* __restrict__ noise, const int* __restrict__ acts,
    const float* __restrict__ b_gru, const float* __restrict__ b_mu,
    const float* __restrict__ b_std, const float* __restrict__ b_r1,
    const float* __restrict__ w_r2, const float* __restrict__ b_r2,
    const bf16_t* __restrict__ pWxs, const bf16_t* __restrict__ pWh,
    const bf16_t* __restrict__ pWmu, const bf16_t* __restrict__ pWstd,
    const bf16_t* __restrict__ pWr1, const float* __restrict__ Wxa,
    float* __restrict__ partials) {
  extern __shared__ __align__(16) char smem[];
  bf16_t* hbuf = (bf16_t*)smem;                           // [2][32][HS] bf16
  bf16_t* sb   = (bf16_t*)(smem + 2 * 32 * HS * 2);       // [32][SS]  bf16
  float*  nbuf = (float*)(smem + 2 * 32 * HS * 2 + 32 * SS * 2);  // [32][SS] f32
  float* reward_acc = (float*)(smem + 2 * 32 * HS * 2 + 32 * SS * 2 + 32 * SS * 4);
  float* sum_acc = reward_acc + 32;
  float* sq_acc  = sum_acc + 32;
  float* efe_acc = sq_acc + 32;

  const int row0 = blockIdx.x * 32;
  const int lane = threadIdx.x & 31;
  const int wave = threadIdx.x >> 5;   // 0..7: unique column stripe

  // Init: broadcast h0/s0 to all 32 rows; zero accumulators.
  for (int i = threadIdx.x; i < 32 * HS; i += 256)
    hbuf[i] = (bf16_t)h0[i & (HS - 1)];
  for (int i = threadIdx.x; i < 32 * SS; i += 256)
    sb[i] = (bf16_t)s0[i & (SS - 1)];
  if (threadIdx.x < 32) {
    reward_acc[threadIdx.x] = 0.f; sum_acc[threadIdx.x] = 0.f;
    sq_acc[threadIdx.x] = 0.f;     efe_acc[threadIdx.x] = 0.f;
  }
  __syncthreads();

  bf16_t* Hc = hbuf;                // current h
  bf16_t* Hn = hbuf + 32 * HS;      // next h
  float disc = 1.f;

  for (int t = 0; t < T_HOR; ++t) {
    // ---- stage noise[t] for this WG's rows: async global->LDS, overlaps A
    {
      const char* gsrc = (const char*)(noise + ((size_t)t * BATCH + row0) * SS);
      const uint32_t lbase = (uint32_t)(uintptr_t)nbuf;
#pragma unroll
      for (int k = 0; k < 8; ++k) {               // 2048 * 16B = 32KB
        const uint32_t off = (uint32_t)(threadIdx.x + k * 256) * 16u;
        async_g2l_b128(lbase + off, (uint64_t)(uintptr_t)(gsrc + off));
      }
    }

    // ---------- Phase A: GRU gates + h update --------------------------
    for (int i = 0; i < 8; ++i) {
      const int cb = wave + 8 * i;    // 0..63: H column block (unique per wave)
      v8f az[2] = {{}, {}}, ar[2] = {{}, {}}, anx[2] = {{}, {}}, anh[2] = {{}, {}};
#pragma unroll 2
      for (int kt = 0; kt < 8; ++kt) {            // s @ Wxs (K=256)
        const v16bf bz = load_B_g(pWxs, kt, cb,       8, lane);
        const v16bf br = load_B_g(pWxs, kt, cb + 64,  8, lane);
        const v16bf bn = load_B_g(pWxs, kt, cb + 128, 8, lane);
        const v16bf a0 = load_A_lds(sb, SS, 0,  kt * 32, lane);
        const v16bf a1 = load_A_lds(sb, SS, 16, kt * 32, lane);
        az[0] = wmma_bf16(a0, bz, az[0]);  az[1] = wmma_bf16(a1, bz, az[1]);
        ar[0] = wmma_bf16(a0, br, ar[0]);  ar[1] = wmma_bf16(a1, br, ar[1]);
        anx[0] = wmma_bf16(a0, bn, anx[0]); anx[1] = wmma_bf16(a1, bn, anx[1]);
      }
#pragma unroll 2
      for (int kt = 0; kt < 32; ++kt) {           // h @ Wh (K=1024)
        if (kt < 31)
          __builtin_prefetch(pWh + ((size_t)(cb * 32 + kt + 1)) * 512, 0, 1);
        const v16bf bz = load_B_g(pWh, kt, cb,       32, lane);
        const v16bf br = load_B_g(pWh, kt, cb + 64,  32, lane);
        const v16bf bn = load_B_g(pWh, kt, cb + 128, 32, lane);
        const v16bf a0 = load_A_lds(Hc, HS, 0,  kt * 32, lane);
        const v16bf a1 = load_A_lds(Hc, HS, 16, kt * 32, lane);
        az[0] = wmma_bf16(a0, bz, az[0]);  az[1] = wmma_bf16(a1, bz, az[1]);
        ar[0] = wmma_bf16(a0, br, ar[0]);  ar[1] = wmma_bf16(a1, br, ar[1]);
        anh[0] = wmma_bf16(a0, bn, anh[0]); anh[1] = wmma_bf16(a1, bn, anh[1]);
      }
      const int n = lane & 15;
      const int col = cb * 16 + n;
      const float bzv = b_gru[col], brv = b_gru[col + HS], bnv = b_gru[col + 2 * HS];
      const int hi = (lane >> 4) << 3;
#pragma unroll
      for (int r2 = 0; r2 < 2; ++r2)
#pragma unroll
        for (int v = 0; v < 8; ++v) {             // C-tile: row = v + hi
          const int m = r2 * 16 + v + hi;
          const int act = acts[t * BATCH + row0 + m];
          const float* wa = Wxa + (size_t)act * 3 * HS;  // one-hot -> bias row
          const float z = 1.f / (1.f + __expf(-(az[r2][v] + wa[col] + bzv)));
          const float r = 1.f / (1.f + __expf(-(ar[r2][v] + wa[col + HS] + brv)));
          const float nn = tanhf(anx[r2][v] + wa[col + 2 * HS] + r * anh[r2][v] + bnv);
          const float hold = (float)Hc[m * HS + col];
          Hn[m * HS + col] = (bf16_t)((1.f - z) * hold + z * nn);
        }
    }
    wait_asynccnt0();   // noise tile resident before the barrier
    __syncthreads();

    // ---------- Phase B: latent sample s_new ---------------------------
    for (int i = 0; i < 2; ++i) {
      const int cb = wave * 2 + i;    // 0..15: S column block (unique per wave)
      v8f amu[2] = {{}, {}}, asd[2] = {{}, {}};
#pragma unroll 2
      for (int kt = 0; kt < 32; ++kt) {
        const v16bf bm = load_B_g(pWmu,  kt, cb, 32, lane);
        const v16bf bs = load_B_g(pWstd, kt, cb, 32, lane);
        const v16bf a0 = load_A_lds(Hn, HS, 0,  kt * 32, lane);
        const v16bf a1 = load_A_lds(Hn, HS, 16, kt * 32, lane);
        amu[0] = wmma_bf16(a0, bm, amu[0]); amu[1] = wmma_bf16(a1, bm, amu[1]);
        asd[0] = wmma_bf16(a0, bs, asd[0]); asd[1] = wmma_bf16(a1, bs, asd[1]);
      }
      const int n = lane & 15;
      const int col = cb * 16 + n;
      const float bm = b_mu[col], bs = b_std[col];
      const int hi = (lane >> 4) << 3;
#pragma unroll
      for (int r2 = 0; r2 < 2; ++r2)
#pragma unroll
        for (int v = 0; v < 8; ++v) {
          const int m = r2 * 16 + v + hi;
          const float stdv = log1pf(__expf(asd[r2][v] + bs)) + 0.1f; // softplus+0.1
          const float eps = nbuf[m * SS + col];
          sb[m * SS + col] = (bf16_t)(amu[r2][v] + bm + stdv * eps);
        }
    }
    __syncthreads();

    // ---------- Phase V: variance sums over feat = [h_new, s_new] ------
    {
      const int rw = threadIdx.x >> 3;  // row 0..31
      const int pp = threadIdx.x & 7;   // 8-way split of 1280 features
      float sm = 0.f, sq = 0.f;
      const bf16_t* hr = Hn + rw * HS + pp * 128;
      for (int j = 0; j < 128; ++j) { const float x = (float)hr[j]; sm += x; sq += x * x; }
      const bf16_t* sr = sb + rw * SS + pp * 32;
      for (int j = 0; j < 32; ++j)  { const float x = (float)sr[j]; sm += x; sq += x * x; }
      atomicAdd(&sum_acc[rw], sm);
      atomicAdd(&sq_acc[rw], sq);
    }

    // ---------- Phase C: reward MLP ------------------------------------
    for (int i = 0; i < 4; ++i) {
      const int cb = wave + 8 * i;     // 0..31: RH column block (unique per wave)
      v8f acc[2] = {{}, {}};
#pragma unroll 2
      for (int kt = 0; kt < 32; ++kt) {           // h part of feat (K=1024)
        const v16bf b = load_B_g(pWr1, kt, cb, 40, lane);
        const v16bf a0 = load_A_lds(Hn, HS, 0,  kt * 32, lane);
        const v16bf a1 = load_A_lds(Hn, HS, 16, kt * 32, lane);
        acc[0] = wmma_bf16(a0, b, acc[0]); acc[1] = wmma_bf16(a1, b, acc[1]);
      }
#pragma unroll 2
      for (int kt = 0; kt < 8; ++kt) {            // s part of feat (K=256)
        const v16bf b = load_B_g(pWr1, 32 + kt, cb, 40, lane);
        const v16bf a0 = load_A_lds(sb, SS, 0,  kt * 32, lane);
        const v16bf a1 = load_A_lds(sb, SS, 16, kt * 32, lane);
        acc[0] = wmma_bf16(a0, b, acc[0]); acc[1] = wmma_bf16(a1, b, acc[1]);
      }
      const int n = lane & 15;
      const int col = cb * 16 + n;
      const float w2 = w_r2[col], b1 = b_r1[col];
      const int hi = (lane >> 4) << 3;
#pragma unroll
      for (int r2 = 0; r2 < 2; ++r2)
#pragma unroll
        for (int v = 0; v < 8; ++v) {
          const int m = r2 * 16 + v + hi;
          const float hv = fmaxf(acc[r2][v] + b1, 0.f);
          atomicAdd(&reward_acc[m], hv * w2);     // relu(.)@w_r2
        }
    }
    __syncthreads();

    // ---------- step finalize: efe += gamma^t * (reward + var) ---------
    if (threadIdx.x < 32) {
      const int m = threadIdx.x;
      const float mean = sum_acc[m] * (1.f / FEATN);
      const float var = sq_acc[m] * (1.f / FEATN) - mean * mean;
      const float rew = reward_acc[m] + b_r2[0];
      efe_acc[m] += disc * (rew + var);
      sum_acc[m] = 0.f; sq_acc[m] = 0.f; reward_acc[m] = 0.f;
    }
    disc *= 0.99f;
    { bf16_t* tmp = Hc; Hc = Hn; Hn = tmp; }
    __syncthreads();
  }

  if (threadIdx.x == 0) {
    float s = 0.f;
    for (int m = 0; m < 32; ++m) s += efe_acc[m];
    partials[blockIdx.x] = s;   // 32-row partial sum of (pragmatic+epistemic)
  }
}

// 8 WGs per action (256 rows / 32); reduce and negate for EFE mean.
__global__ void efe_finalize(const float* __restrict__ partials,
                             float* __restrict__ out) {
  const int a = threadIdx.x;
  if (a < A_DIM) {
    float s = 0.f;
    for (int j = 0; j < 8; ++j) s += partials[a * 8 + j];
    out[a] = -s * (1.f / NSAMP);
  }
}

// ---------------------------------------------------------------------------
extern "C" void kernel_launch(void* const* d_in, const int* in_sizes, int n_in,
                              void* d_out, int out_size, void* d_ws, size_t ws_size,
                              hipStream_t stream) {
  (void)in_sizes; (void)n_in; (void)out_size; (void)ws_size;
  const float* h0    = (const float*)d_in[0];
  const float* s0    = (const float*)d_in[1];
  const float* noise = (const float*)d_in[2];
  const float* Wx    = (const float*)d_in[3];
  const float* Wh    = (const float*)d_in[4];
  const float* b_gru = (const float*)d_in[5];
  const float* W_mu  = (const float*)d_in[6];
  const float* b_mu  = (const float*)d_in[7];
  const float* W_std = (const float*)d_in[8];
  const float* b_std = (const float*)d_in[9];
  const float* W_r1  = (const float*)d_in[10];
  const float* b_r1  = (const float*)d_in[11];
  const float* w_r2  = (const float*)d_in[12];
  const float* b_r2  = (const float*)d_in[13];
  const int*   acts  = (const int*)d_in[14];

  char* ws = (char*)d_ws;                       // ~10.4 MB used
  bf16_t* pWxs  = (bf16_t*)(ws + 0);            // 256x3072   -> 1.5 MB
  bf16_t* pWh   = (bf16_t*)(ws + 1572864);      // 1024x3072  -> 6.0 MB
  bf16_t* pWmu  = (bf16_t*)(ws + 7864320);      // 1024x256
  bf16_t* pWstd = (bf16_t*)(ws + 8388608);      // 1024x256
  bf16_t* pWr1  = (bf16_t*)(ws + 8912896);      // 1280x512
  float*  Wxa   = (float*)(ws + 10223616);      // 16x3072 fp32 action bias
  float*  part  = (float*)(ws + 10420224);      // 128 partials

  auto pack = [&](const float* W, int K, int N, bf16_t* outp) {
    const int total = (K / 32) * (N / 16) * 32;
    pack_b_kernel<<<(total + 255) / 256, 256, 0, stream>>>(W, K, N, outp);
  };
  pack(Wx,    256, 3072, pWxs);   // first 256 rows of Wx (the s-part)
  pack(Wh,   1024, 3072, pWh);
  pack(W_mu, 1024,  256, pWmu);
  pack(W_std,1024,  256, pWstd);
  pack(W_r1, 1280,  512, pWr1);   // feat order [h, s] matches W_r1 rows
  copy_wxa_kernel<<<(A_DIM * 3 * HS + 255) / 256, 256, 0, stream>>>(Wx, Wxa);

  // LDS: h dbl-buf 128K + s 16K + noise 32K + 4*32 f32 accs = 180,736 B
  const size_t smem = 2 * 32 * HS * 2 + 32 * SS * 2 + 32 * SS * 4 + 4 * 32 * 4;
  efe_rollout<<<BATCH / 32, 256, smem, stream>>>(
      h0, s0, noise, acts, b_gru, b_mu, b_std, b_r1, w_r2, b_r2,
      pWxs, pWh, pWmu, pWstd, pWr1, Wxa, part);
  efe_finalize<<<1, 32, 0, stream>>>(part, (float*)d_out);
}
